// MultiHeadAttention_63230508531953
// MI455X (gfx1250) — matrix operand order repaired and checked by reading, not compiled, not run on hardware
//
#include <hip/hip_runtime.h>
#include <stdint.h>

#define D_MODEL 1024
#define N_HEADS 16
#define HEAD_DIM 64
#define BATCH 4
#define SEQ 2048
#define M_TOTAL (BATCH * SEQ)   // 8192

typedef __bf16 bf16;
typedef __attribute__((ext_vector_type(16))) __bf16 v16bf;
typedef __attribute__((ext_vector_type(8)))  float  v8f;
typedef __attribute__((ext_vector_type(4)))  uint32_t u32x4;
typedef __attribute__((ext_vector_type(8)))  int      i32x8;
typedef __attribute__((ext_vector_type(4)))  int      i32x4;

union ABFrag { v16bf v; bf16 e[16]; };

// -------------------------------------------------------------------------
// GEMM: C[M,N] = A[M,K] * B[N,K]^T + bias[N]
// Block tile 64(M) x 256(N), K-step 32; 8 waves, wave grid 2x4;
// per wave 2x4 accumulators of 16x16 via v_wmma_f32_16x16x32_bf16.
// -------------------------------------------------------------------------
constexpr int BM = 64, BN = 256, BK = 32, PAD = 8;

template<bool A_BF16, bool OUT_BF16>
__global__ __launch_bounds__(256) void gemm_bias_wmma(
    const void* __restrict__ Ap,      // [M][K] f32 or bf16
    const float* __restrict__ Bw,     // [N][K] f32 (K-contiguous weights)
    const float* __restrict__ bias,   // [N]
    void* __restrict__ Cp,            // [M][N] f32 or bf16
    int M, int N, int K)
{
  __shared__ bf16 lA[BM][BK + PAD];
  __shared__ bf16 lB[BN][BK + PAD];

  const int tid   = threadIdx.x;
  const int lane  = tid & 31;
  const int wave  = tid >> 5;
  const int waveM = wave >> 2;          // 0..1
  const int waveN = wave & 3;           // 0..3
  const int bm = blockIdx.y * BM;
  const int bn = blockIdx.x * BN;

  v8f acc[2][4] = {};

  const int aRow = tid >> 2;            // 0..63
  const int aCol = (tid & 3) * 8;       // 0,8,16,24
  const int bRow = tid;                 // 0..255

  for (int kk = 0; kk < K; kk += BK) {
    if constexpr (A_BF16) {
      const bf16* Ag = (const bf16*)Ap + (size_t)(bm + aRow) * K + kk + aCol;
      #pragma unroll
      for (int i = 0; i < 8; ++i) lA[aRow][aCol + i] = Ag[i];
    } else {
      const float* Ag = (const float*)Ap + (size_t)(bm + aRow) * K + kk + aCol;
      #pragma unroll
      for (int i = 0; i < 8; ++i) lA[aRow][aCol + i] = (bf16)Ag[i];
    }
    const float* Bg = Bw + (size_t)(bn + bRow) * K + kk;
    #pragma unroll
    for (int i = 0; i < 32; ++i) lB[bRow][i] = (bf16)Bg[i];
    if (kk + BK < K) __builtin_prefetch((const void*)(Bg + BK), 0, 1);
    __syncthreads();

    const int nrow = lane & 15;
    const int krow = (lane >> 4) << 3;    // A frag: K base 0/8
    ABFrag af[2];
    #pragma unroll
    for (int mt = 0; mt < 2; ++mt) {
      const int r = waveM * 32 + mt * 16 + nrow;
      #pragma unroll
      for (int i = 0; i < 16; ++i)
        af[mt].e[i] = lA[r][krow + (i & 7) + ((i >> 3) << 4)];
    }
    const int kcol = (lane >> 4) << 4;    // unused; B frag uses krow (K base 0/8)
    #pragma unroll
    for (int nt = 0; nt < 4; ++nt) {
      ABFrag bfr;
      const int c = waveN * 64 + nt * 16 + nrow;
      #pragma unroll
      for (int i = 0; i < 16; ++i) bfr.e[i] = lB[c][krow + (i & 7) + ((i >> 3) << 4)];
      #pragma unroll
      for (int mt = 0; mt < 2; ++mt)
        acc[mt][nt] = __builtin_amdgcn_wmma_f32_16x16x32_bf16(
            false, af[mt].v, false, bfr.v, (short)0, acc[mt][nt], false, false);
    }
    __syncthreads();
  }

  #pragma unroll
  for (int nt = 0; nt < 4; ++nt) {
    const int n = bn + waveN * 64 + nt * 16 + (lane & 15);
    const float bn_v = bias[n];
    #pragma unroll
    for (int mt = 0; mt < 2; ++mt)
      #pragma unroll
      for (int i = 0; i < 8; ++i) {
        const int m = bm + waveM * 32 + mt * 16 + ((lane >> 4) << 3) + i;
        const float v = acc[mt][nt][i] + bn_v;
        if constexpr (OUT_BF16) ((bf16*)Cp)[(size_t)m * N + n] = (bf16)v;
        else                    ((float*)Cp)[(size_t)m * N + n] = v;
      }
  }
}

// -------------------------------------------------------------------------
// Flash attention, one workgroup per (b,h) x 128 query rows, 16 rows/wave.
// K+V (adjacent 128 bf16 per row) staged via the Tensor Data Mover into LDS.
// S  = Q K^T    (D layout: M=query, N=key)
// O^T = V^T P^T (D layout: M=feat,  N=query)  -> softmax state is per-lane.
// -------------------------------------------------------------------------
constexpr int KV_CHUNK = 32;
constexpr int SPAD = 4;

__global__ __launch_bounds__(256) void flash_attn_wmma(
    const bf16* __restrict__ qkv,      // [B*T][3*D_MODEL]
    bf16* __restrict__ attn_out)       // [B*T][D_MODEL] head-major features
{
  __shared__ bf16  lKV[KV_CHUNK][2 * HEAD_DIM];        // [key][0:63]=K, [64:127]=V
  __shared__ float lPS[8][16][KV_CHUNK + SPAD];        // per-wave raw S scores

  const int tid = threadIdx.x, lane = tid & 31, wave = tid >> 5;
  const int b = blockIdx.x / N_HEADS, h = blockIdx.x % N_HEADS;
  const int qbase = blockIdx.y * 128 + wave * 16;

  const size_t rowStride = 3 * D_MODEL;
  const bf16* qkv_b = qkv + (size_t)b * SEQ * rowStride + (size_t)h * (3 * HEAD_DIM);

  // Q fragments (A layout, M=query): head dim 64 = two K=32 chunks.
  ABFrag qa[2];
  {
    const int m = lane & 15;
    const int krow = (lane >> 4) << 3;
    const bf16* qr = qkv_b + (size_t)(qbase + m) * rowStride;
    #pragma unroll
    for (int c = 0; c < 2; ++c)
      #pragma unroll
      for (int i = 0; i < 16; ++i)
        qa[c].e[i] = qr[c * 32 + krow + (i & 7) + ((i >> 3) << 4)];
  }

  v8f oacc[4] = {};                 // O^T tiles: M=feat (4x16), N=query
  float rmax = -3.0e38f, rsum = 0.f;
  const float scale = 0.125f;       // hd^-0.5

  // Constant part of the TDM descriptor (group 1): data_size=2B,
  // tensor_dim0=128 (K|V slice width), dim0 stride = 3072 elements,
  // tile = 128 x 32 rows.
  i32x8 g1;
  g1[0] = 1 << 16;                                  // data_size = 2 bytes
  g1[1] = (int)((128u & 0xFFFFu) << 16);            // tensor_dim0[15:0]
  g1[2] = (int)((SEQ & 0xFFFFu) << 16);             // tensor_dim0 hi=0 | tensor_dim1 lo
  g1[3] = (int)(128u << 16);                        // tensor_dim1 hi=0 | tile_dim0=128
  g1[4] = (int)(KV_CHUNK & 0xFFFFu);                // tile_dim1=32, tile_dim2=0
  g1[5] = (int)(3 * D_MODEL);                       // tensor_dim0_stride[31:0]
  g1[6] = 0;                                        // stride hi | dim1_stride lo
  g1[7] = 0;
  const i32x4 gz4 = {0, 0, 0, 0};
  const i32x8 gz8 = {0, 0, 0, 0, 0, 0, 0, 0};

  for (int kv = 0; kv < SEQ; kv += KV_CHUNK) {
    // ---- TDM: DMA the 32x128 bf16 K|V tile into LDS (wave 0 only) ----
    if (wave == 0) {
      const uint64_t ga = (uint64_t)(uintptr_t)(qkv_b + (size_t)kv * rowStride + HEAD_DIM);
      u32x4 g0;
      g0[0] = 1u;                                              // count=1
      g0[1] = (uint32_t)(uintptr_t)(&lKV[0][0]);               // lds_addr
      g0[2] = (uint32_t)ga;                                    // global_addr lo
      g0[3] = (uint32_t)((ga >> 32) & 0x01FFFFFFu) | (2u << 30); // addr hi | type=2
      __builtin_amdgcn_tensor_load_to_lds(g0, g1, gz4, gz4, gz8, 0);
      __builtin_amdgcn_s_wait_tensorcnt(0);
    }
    __syncthreads();

    // ---- S = Q K^T : two 16x16 key tiles, K(=hd) split into 2 WMMAs ----
    v8f s[2] = {};
    #pragma unroll
    for (int t2 = 0; t2 < 2; ++t2) {
      const int key = t2 * 16 + (lane & 15);
      const int kc  = (lane >> 4) << 3;
      #pragma unroll
      for (int c = 0; c < 2; ++c) {
        ABFrag kb;
        #pragma unroll
        for (int i = 0; i < 16; ++i) kb.e[i] = lKV[key][c * 32 + kc + (i & 7) + ((i >> 3) << 4)];
        s[t2] = __builtin_amdgcn_wmma_f32_16x16x32_bf16(
            false, qa[c].v, false, kb.v, (short)0, s[t2], false, false);
      }
    }

    // ---- stage scaled scores, re-read one query row per lane ----
    #pragma unroll
    for (int t2 = 0; t2 < 2; ++t2)
      #pragma unroll
      for (int i = 0; i < 8; ++i)
        lPS[wave][((lane >> 4) << 3) + i][t2 * 16 + (lane & 15)] = s[t2][i] * scale;
    asm volatile("s_wait_dscnt 0" ::: "memory");   // per-wave LDS transpose RAW

    const int rq  = lane & 15;            // query owned by this lane
    const int kb2 = (lane >> 4) << 4;     // this lane's 16-key span
    float vals[16];
    #pragma unroll
    for (int j = 0; j < 16; ++j) vals[j] = lPS[wave][rq][kb2 + j];

    // ---- online softmax: pure VALU + ONE xor-16 shuffle per reduction ----
    float mx = vals[0];
    #pragma unroll
    for (int j = 1; j < 16; ++j) mx = fmaxf(mx, vals[j]);
    mx = fmaxf(mx, __shfl_xor(mx, 16, 32));
    const float nm = fmaxf(rmax, mx);
    const float corr = __expf(rmax - nm);
    rmax = nm;

    ABFrag pb;                            // P^T B-fragment, built in registers
    float ps = 0.f;
    #pragma unroll
    for (int j = 0; j < 16; ++j) {
      const float p = __expf(vals[j] - nm);
      ps += p;
      pb.e[j] = (bf16)p;
    }
    ps += __shfl_xor(ps, 16, 32);
    rsum = rsum * corr + ps;

    #pragma unroll
    for (int nt = 0; nt < 4; ++nt)
      #pragma unroll
      for (int i = 0; i < 8; ++i) oacc[nt][i] *= corr;   // corr is per-query=per-lane

    // ---- O^T += V^T P^T : A = V^T (feat x key), B = P^T (key x query) ----
    const int krow = (lane >> 4) << 3;
    #pragma unroll
    for (int ft = 0; ft < 4; ++ft) {
      ABFrag va;
      const int feat = ft * 16 + (lane & 15);
      #pragma unroll
      for (int i = 0; i < 16; ++i) {
        const int key = ((lane >> 4) << 4) + i;
        va.e[i] = lKV[key][HEAD_DIM + feat];
      }
      oacc[ft] = __builtin_amdgcn_wmma_f32_16x16x32_bf16(
          false, va.v, false, pb.v, (short)0, oacc[ft], false, false);
    }
    __syncthreads();
  }

  // ---- normalize and store: lane owns query=lane%16, 8 consecutive feats ----
  const float inv = 1.f / rsum;
  bf16* orow = attn_out + (size_t)((size_t)b * SEQ + qbase + (lane & 15)) * D_MODEL
             + (size_t)h * HEAD_DIM;
  #pragma unroll
  for (int ft = 0; ft < 4; ++ft)
    #pragma unroll
    for (int i = 0; i < 8; ++i) {
      const int feat = ft * 16 + ((lane >> 4) << 3) + i;
      orow[feat] = (bf16)(oacc[ft][i] * inv);
    }
}

// -------------------------------------------------------------------------
extern "C" void kernel_launch(void* const* d_in, const int* in_sizes, int n_in,
                              void* d_out, int out_size, void* d_ws, size_t ws_size,
                              hipStream_t stream) {
  const float* x    = (const float*)d_in[0];
  const float* Wqkv = (const float*)d_in[1];
  const float* bqkv = (const float*)d_in[2];
  const float* Wo   = (const float*)d_in[3];
  const float* bo   = (const float*)d_in[4];
  float* out = (float*)d_out;

  bf16* qkv  = (bf16*)d_ws;                              // 8192*3072 bf16 (48 MB)
  bf16* attn = qkv + (size_t)M_TOTAL * 3 * D_MODEL;      // 8192*1024 bf16 (16 MB)

  dim3 blk(256);
  gemm_bias_wmma<false, true><<<dim3((3 * D_MODEL) / BN, M_TOTAL / BM), blk, 0, stream>>>(
      x, Wqkv, bqkv, qkv, M_TOTAL, 3 * D_MODEL, D_MODEL);
  flash_attn_wmma<<<dim3(BATCH * N_HEADS, SEQ / 128), blk, 0, stream>>>(qkv, attn);
  gemm_bias_wmma<true, false><<<dim3(D_MODEL / BN, M_TOTAL / BM), blk, 0, stream>>>(
      attn, Wo, bo, out, M_TOTAL, D_MODEL, D_MODEL);
}
